// DeformableConv1D_3152505995722
// MI455X (gfx1250) — matrix operand order repaired and checked by reading, not compiled, not run on hardware
//
#include <hip/hip_runtime.h>
#include <stdint.h>

// ---- problem constants (match reference) ----
#define B_    32
#define CIN   64
#define COUT  128
#define KW    3
#define WDIM  8192
// ---- tiling ----
#define WT    64                 // output columns per workgroup
#define HALO  8
#define XCOLS (WT + 2*HALO)      // 80 staged x columns (20 x 16B chunks)
#define XSTR  (XCOLS + 4)        // 84 floats = 336 B row stride (16B aligned)
#define CK    (KW*CIN)           // 192 = GEMM reduction length
#define CKSTR 200                // padded bf16 row stride (400 B, 16B aligned)

// ---- dynamic LDS layout (bytes, all 16B aligned) ----
#define OFF_XS   0                         // float [CIN][XSTR]      = 21504 B
#define OFF_WOF  21504                     // float [6*CIN*KW]       =  4608 B
#define OFF_BD   26112                     // float [COUT]           =   512 B
#define OFF_A0   26624                     // float [KW*WT]          =   768 B
#define OFF_A1   27392                     // float [KW*WT]          =   768 B
#define OFF_I0   28160                     // int   [KW*WT]          =   768 B
#define OFF_I1   28928                     // int   [KW*WT]          =   768 B
#define OFF_WD   29696                     // bf16  [COUT][CKSTR]    = 51200 B
#define OFF_SP   80896                     // bf16  [WT][CKSTR]      = 25600 B
#define SMEM_BYTES 106496

typedef __attribute__((ext_vector_type(16))) __bf16   v16bf;
typedef __attribute__((ext_vector_type(8)))  float    v8f;
typedef __attribute__((ext_vector_type(4)))  uint32_t u32x4;
typedef __attribute__((ext_vector_type(4)))  int      int4v;

#if __has_builtin(__builtin_amdgcn_global_load_async_to_lds_b128)
#define HAVE_ASYNC_LDS 1
typedef __attribute__((address_space(1))) int4v GInt4;  // global int4
typedef __attribute__((address_space(3))) int4v LInt4;  // LDS int4
#endif

__global__ __launch_bounds__(256)
void deform_conv1d_wmma(const float* __restrict__ x,
                        const float* __restrict__ w_off,
                        const float* __restrict__ b_off,
                        const float* __restrict__ w_def,
                        const float* __restrict__ b_def,
                        float* __restrict__ out)
{
    extern __shared__ char smem[];
    float*  xs   = (float*)(smem + OFF_XS);
    float*  wofs = (float*)(smem + OFF_WOF);
    float*  bd   = (float*)(smem + OFF_BD);
    float*  a0s  = (float*)(smem + OFF_A0);
    float*  a1s  = (float*)(smem + OFF_A1);
    int*    i0s  = (int*)  (smem + OFF_I0);
    int*    i1s  = (int*)  (smem + OFF_I1);
    __bf16* wd   = (__bf16*)(smem + OFF_WD);
    __bf16* sp   = (__bf16*)(smem + OFF_SP);

    const int tid  = threadIdx.x;
    const int wave = tid >> 5;
    const int lane = tid & 31;
    const int b    = blockIdx.x / (WDIM / WT);
    const int w0   = (blockIdx.x % (WDIM / WT)) * WT;

    // ---- stage x tile with halo ----
#ifdef HAVE_ASYNC_LDS
    const bool interior = (w0 >= HALO) && (w0 + WT + HALO <= WDIM);
    if (interior) {
        // Direct global -> LDS DMA, 16B per lane-op, tracked by ASYNCcnt.
        // 64 rows x 20 chunks = 1280 ops = exactly 5 per thread.
        for (int idx = tid; idx < CIN * (XCOLS / 4); idx += 256) {
            int c = idx / (XCOLS / 4);
            int q = idx % (XCOLS / 4);
            const float* gp = &x[(b * CIN + c) * WDIM + (w0 - HALO) + q * 4];
            float*       lp = &xs[c * XSTR + q * 4];
            __builtin_amdgcn_global_load_async_to_lds_b128(
                (GInt4*)(void*)gp, (LInt4*)(void*)lp, 0, 0);
        }
        asm volatile("s_wait_asynccnt 0" ::: "memory");
    } else
#endif
    {
        for (int idx = tid; idx < CIN * XCOLS; idx += 256) {
            int c = idx / XCOLS, col = idx % XCOLS;
            int g = w0 - HALO + col;
            float v = 0.0f;
            if (g >= 0 && g < WDIM) v = x[(b * CIN + c) * WDIM + g];
            xs[c * XSTR + col] = v;
        }
    }
    // ---- stage offset-conv weights ----
    for (int idx = tid; idx < 2 * KW * CIN * KW; idx += 256) wofs[idx] = w_off[idx];
    // ---- stage output bias ----
    if (tid < COUT) bd[tid] = b_def[tid];
    // ---- stage w_def as bf16, layout [o][ck], ck = k*CIN + c ----
    for (int idx = tid; idx < COUT * CK; idx += 256) {
        int o = idx / CK, ck = idx % CK;
        int k = ck / CIN, c = ck % CIN;
        wd[o * CKSTR + ck] = (__bf16)w_def[(o * CIN + c) * KW + k];
    }
    __syncthreads();

    // ---- offset conv + sampling params, one thread per (k, w) ----
    if (tid < KW * WT) {
        int k  = tid / WT;
        int wl = tid % WT;
        float dy = b_off[2 * k];
        float dx = b_off[2 * k + 1];
        const float* wr0 = &wofs[(2 * k)     * CIN * KW];
        const float* wr1 = &wofs[(2 * k + 1) * CIN * KW];
        for (int c = 0; c < CIN; ++c) {
            float xv0 = xs[c * XSTR + wl + HALO - 1];
            float xv1 = xs[c * XSTR + wl + HALO    ];
            float xv2 = xs[c * XSTR + wl + HALO + 1];
            dy += wr0[c * KW + 0] * xv0 + wr0[c * KW + 1] * xv1 + wr0[c * KW + 2] * xv2;
            dx += wr1[c * KW + 0] * xv0 + wr1[c * KW + 1] * xv1 + wr1[c * KW + 2] * xv2;
        }
        float wy  = fmaxf(0.0f, 1.0f - fabsf(dy));
        float pos = (float)(w0 + wl - 1 + k) + dx;
        float x0f = floorf(pos);
        float wx1 = pos - x0f;
        int   x0  = (int)x0f;
        int   x1  = x0 + 1;
        float v0 = (x0 >= 0 && x0 < WDIM) ? 1.0f : 0.0f;
        float v1 = (x1 >= 0 && x1 < WDIM) ? 1.0f : 0.0f;
        a0s[tid] = wy * (1.0f - wx1) * v0;
        a1s[tid] = wy * wx1 * v1;
        int il0 = x0 - (w0 - HALO); il0 = il0 < 0 ? 0 : (il0 > XCOLS - 1 ? XCOLS - 1 : il0);
        int il1 = x1 - (w0 - HALO); il1 = il1 < 0 ? 0 : (il1 > XCOLS - 1 ? XCOLS - 1 : il1);
        i0s[tid] = il0;
        i1s[tid] = il1;
    }
    __syncthreads();

    // ---- build samp tile in bf16: sp[w][ck] (B-matrix staging) ----
    for (int idx = tid; idx < WT * CK; idx += 256) {
        int ck = idx % CK, w = idx / CK;
        int k = ck / CIN, c = ck % CIN;
        int p = k * WT + w;
        float v = a0s[p] * xs[c * XSTR + i0s[p]] + a1s[p] * xs[c * XSTR + i1s[p]];
        sp[w * CKSTR + ck] = (__bf16)v;
    }
    __syncthreads();

    // ---- WMMA GEMM: each wave = one 16-row co tile x four 16-col w tiles ----
    v8f acc[4];
    #pragma unroll
    for (int nt = 0; nt < 4; ++nt) acc[nt] = (v8f){0.f,0.f,0.f,0.f,0.f,0.f,0.f,0.f};

    const int half = lane >> 4;       // K-half selector per fragment layouts
    const int l16  = lane & 15;
    const int orow = wave * 16 + l16; // A-matrix row (co) for this lane

    union Frag { v16bf v; u32x4 q[2]; };

    #pragma unroll
    for (int chunk = 0; chunk < CK / 32; ++chunk) {
        const int ck0 = chunk * 32;
        // Load group first (A + all 4 B fragments), then compute group,
        // so WMMAs can issue back-to-back after one DS drain.
        Frag A;
        // 16-bit A 16x32 layout: lanes0-15 K={0..7,16..23}, lanes16-31 K={8..15,24..31}
        A.q[0] = *(const u32x4*)&wd[orow * CKSTR + ck0 +      8 * half];
        A.q[1] = *(const u32x4*)&wd[orow * CKSTR + ck0 + 16 + 8 * half];
        Frag Bt[4];
        #pragma unroll
        for (int nt = 0; nt < 4; ++nt) {
            // 16-bit B 32x16 layout: lane = N col, lanes0-15 K={0..7,16..23}, lanes16-31 K={8..15,24..31}
            const u32x4* pB = (const u32x4*)&sp[(nt * 16 + l16) * CKSTR + ck0 + 8 * half];
            Bt[nt].q[0] = pB[0];
            Bt[nt].q[1] = pB[2];
        }
        #pragma unroll
        for (int nt = 0; nt < 4; ++nt) {
            acc[nt] = __builtin_amdgcn_wmma_f32_16x16x32_bf16(
                false, A.v, false, Bt[nt].v, (short)0, acc[nt], false, false);
        }
    }

    // ---- epilogue: D 16x16 layout -> out[b, o, w] + bias (non-temporal) ----
    #pragma unroll
    for (int nt = 0; nt < 4; ++nt) {
        int wcol = w0 + nt * 16 + l16;
        #pragma unroll
        for (int r = 0; r < 8; ++r) {
            int o = wave * 16 + r + 8 * half;
            __builtin_nontemporal_store(acc[nt][r] + bd[o],
                                        &out[(b * COUT + o) * WDIM + wcol]);
        }
    }
}

extern "C" void kernel_launch(void* const* d_in, const int* in_sizes, int n_in,
                              void* d_out, int out_size, void* d_ws, size_t ws_size,
                              hipStream_t stream) {
    const float* x     = (const float*)d_in[0];
    const float* w_off = (const float*)d_in[1];
    const float* b_off = (const float*)d_in[2];
    const float* w_def = (const float*)d_in[3];
    const float* b_def = (const float*)d_in[4];
    float* out = (float*)d_out;

    dim3 grid(B_ * (WDIM / WT));   // 32 * 128 = 4096 workgroups
    dim3 block(256);               // 8 wave32s
    deform_conv1d_wmma<<<grid, block, SMEM_BYTES, stream>>>(x, w_off, b_off, w_def, b_def, out);
}
